// StructuredLinear_34969623724267
// MI455X (gfx1250) — matrix operand order repaired and checked by reading, not compiled, not run on hardware
//
#include <hip/hip_runtime.h>

typedef _Float16 h8   __attribute__((ext_vector_type(8)));    // 16B
typedef _Float16 v16h __attribute__((ext_vector_type(16)));   // 32B
typedef float    v8f  __attribute__((ext_vector_type(8)));

#define N1    32
#define N2    64
#define M1    32
#define M2    64
#define RANK  4
#define IN_F  2048
#define OUT_F 2048
#define TB    16          // batch rows per workgroup
#define THREADS 512       // 16 wave32s
#define NWAVES  16
#define VC      16        // v values per chunk -> 4 chunks, K/chunk = 64

// LDS layout (_Float16 elements), total 61440 halfs = 120 KB -> 2 WGs/WGP:
//   sXc [16 vv][16 b][32 u]          =  8192   (x chunk, f16, streamed)
//   sW1 [128 j=(m*4+r)][32 u]        =  4096   (core1 transposed)
//   sW2 [64 n][256 k=(v*4+r)]        = 16384   (core2 re-laid)
//   sL  [32 m][16 b][64 kk=(vv*4+r)] = 32768   (stage-1 output, per chunk)
#define OFF_SXC 0
#define OFF_SW1 8192
#define OFF_SW2 12288
#define OFF_SL  28672
#define LDS_HALFS 61440

__global__ __launch_bounds__(THREADS)
void btt_fused_kernel(const float* __restrict__ x,
                      const float* __restrict__ c1,
                      const float* __restrict__ c2,
                      const float* __restrict__ bias,
                      float* __restrict__ out)
{
    extern __shared__ _Float16 smem[];
    _Float16* sXc = smem + OFF_SXC;
    _Float16* sW1 = smem + OFF_SW1;
    _Float16* sW2 = smem + OFF_SW2;
    _Float16* sL  = smem + OFF_SL;

    const int tid  = threadIdx.x;
    const int wave = tid >> 5;
    const int lane = tid & 31;
    const int lrow = lane & 15;   // WMMA M/N position
    const int lhal = lane >> 4;   // lane half (0/1)
    const long tile = blockIdx.x; // one 16-row batch tile per block

    // ---------------- weights -> f16 LDS (consumed after first barrier) ----
    // core1 flat g = (u*32 + m)*4 + r  ->  sW1[(m*4+r)*32 + u]
    for (int g = tid; g < N1 * M1 * RANK; g += THREADS) {
        int r = g & 3, m = (g >> 2) & 31, u = g >> 7;
        sW1[(m * RANK + r) * N1 + u] = (_Float16)c1[g];
    }
    // core2 flat g = (r*64 + v)*64 + n ->  sW2[n*256 + v*4 + r]
    for (int g = tid; g < RANK * N2 * M2; g += THREADS) {
        int n = g & 63, v = (g >> 6) & 63, r = g >> 12;
        sW2[n * 256 + v * RANK + r] = (_Float16)c2[g];
    }

    // ---------------- x streaming: thread t owns (b = t>>5, u = t&31) ------
    // global row segment: x[tile*16 + b][u*64 + v], chunk vc covers v = vc*16..+15
    const float* xg = x + tile * (long)(TB * IN_F) + (tid >> 5) * IN_F + (tid & 31) * 64;
    _Float16* sxp = sXc + (tid >> 5) * 32 + (tid & 31);   // + vv*512
    float4 xr0 = *(const float4*)(xg + 0);
    float4 xr1 = *(const float4*)(xg + 4);
    float4 xr2 = *(const float4*)(xg + 8);
    float4 xr3 = *(const float4*)(xg + 12);

    // -------- hoisted per-wave LDS bases (independent of chunk index) ------
    // stage-1 A: sXc[(wave*16 + lrow)*32 + lhal*16]  (vv = wave)
    const _Float16* a1p = &sXc[(wave * 16 + lrow) * 32 + lhal * 16];
    // stage-1 B: sW1[(jc*16+lrow)*32 + lhal*16]
    const _Float16* b1p = &sW1[lrow * 32 + lhal * 16];
    // stage-1 D store: sL[(m*16 + (q+8*lhal))*64 + wave*4 + r], m=n>>2, r=n&3
    // stage-2 A: sL[((wave*2+ml)*16 + lrow)*64 + s*32 + lhal*16]
    const _Float16* a2p = &sL[(wave * 32 + lrow) * 64 + lhal * 16];
    // stage-2 B: sW2[(nc*16+lrow)*256 + vc*64 + s*32 + lhal*16]
    const _Float16* b2p = &sW2[lrow * 256 + lhal * 16];

    const v8f vzero = {0.f, 0.f, 0.f, 0.f, 0.f, 0.f, 0.f, 0.f};
    v8f acc[8];
#pragma unroll
    for (int i = 0; i < 8; ++i) acc[i] = vzero;

    // ---------------- chunked fused pipeline over v ------------------------
    for (int vc = 0; vc < N2 / VC; ++vc) {
        // spill staged x registers -> sXc[vv][b][u]
        sxp[0 * 512] = (_Float16)xr0.x;  sxp[1 * 512] = (_Float16)xr0.y;
        sxp[2 * 512] = (_Float16)xr0.z;  sxp[3 * 512] = (_Float16)xr0.w;
        sxp[4 * 512] = (_Float16)xr1.x;  sxp[5 * 512] = (_Float16)xr1.y;
        sxp[6 * 512] = (_Float16)xr1.z;  sxp[7 * 512] = (_Float16)xr1.w;
        sxp[8 * 512] = (_Float16)xr2.x;  sxp[9 * 512] = (_Float16)xr2.y;
        sxp[10 * 512] = (_Float16)xr2.z; sxp[11 * 512] = (_Float16)xr2.w;
        sxp[12 * 512] = (_Float16)xr3.x; sxp[13 * 512] = (_Float16)xr3.y;
        sxp[14 * 512] = (_Float16)xr3.z; sxp[15 * 512] = (_Float16)xr3.w;
        __syncthreads();

        // prefetch next chunk's x while stage-1/2 compute (VMEM latency hidden)
        if (vc < N2 / VC - 1) {
            const float* g = xg + (vc + 1) * VC;
            xr0 = *(const float4*)(g + 0);
            xr1 = *(const float4*)(g + 4);
            xr2 = *(const float4*)(g + 8);
            xr3 = *(const float4*)(g + 12);
        }

        // ===== stage 1: this wave handles vv = wave, v = vc*16 + wave ======
        v16h a1;
        {
            h8 lo = *(const h8*)(a1p);
            h8 hi = *(const h8*)(a1p + 8);
            a1 = __builtin_shufflevector(lo, hi, 0,1,2,3,4,5,6,7,8,9,10,11,12,13,14,15);
        }
#pragma unroll
        for (int jp = 0; jp < 4; ++jp) {
            // paired WMMAs with independent D regs: fills WMMA->VALU hazard slots
            const int n0 = (jp * 2) * 16 + lrow;        // j for pair element 0
            v16h bA = *(const v16h*)(b1p + (jp * 2) * 512);
            v16h bB = *(const v16h*)(b1p + (jp * 2 + 1) * 512);
            v8f d0 = __builtin_amdgcn_wmma_f32_16x16x32_f16(
                         false, a1, false, bA, (short)0, vzero, false, false);
            v8f d1 = __builtin_amdgcn_wmma_f32_16x16x32_f16(
                         false, a1, false, bB, (short)0, vzero, false, false);
            // D element q -> (b = q + 8*lhal, j = n); scatter f16 into sL
            const int m0 = n0 >> 2, r0 = n0 & 3;
            _Float16* sp = &sL[(m0 * 16 + lhal * 8) * 64 + wave * 4 + r0];
#pragma unroll
            for (int q = 0; q < 8; ++q) sp[q * 64] = (_Float16)d0[q];
#pragma unroll
            for (int q = 0; q < 8; ++q) sp[4096 + q * 64] = (_Float16)d1[q]; // n0+16 -> m0+4
        }
        __syncthreads();   // sL produced -> consume

        // ===== stage 2: acc[b, m, n] += over k=(vv,r) of this chunk ========
#pragma unroll
        for (int s = 0; s < 2; ++s) {           // K = 64 per chunk -> 2 k-steps
            v16h bf[4];
#pragma unroll
            for (int nc = 0; nc < 4; ++nc)      // B: col n, K = lhal*16 + 0..15
                bf[nc] = *(const v16h*)(b2p + nc * 4096 + vc * 64 + s * 32);
#pragma unroll
            for (int ml = 0; ml < 2; ++ml) {
                const _Float16* ap = a2p + ml * 1024 + s * 32;
                h8 lo = *(const h8*)(ap);
                h8 hi = *(const h8*)(ap + 8);
                v16h a2 = __builtin_shufflevector(lo, hi,
                              0,1,2,3,4,5,6,7,8,9,10,11,12,13,14,15);
#pragma unroll
                for (int nc = 0; nc < 4; ++nc) {
                    acc[ml * 4 + nc] = __builtin_amdgcn_wmma_f32_16x16x32_f16(
                        false, a2, false, bf[nc], (short)0, acc[ml * 4 + nc],
                        false, false);
                }
            }
        }
        __syncthreads();   // sL + sXc consumed -> safe to overwrite next chunk
    }

    // ---------------- epilogue: bias + store fp32 --------------------------
    float* yt = out + tile * (long)(TB * OUT_F);
#pragma unroll
    for (int ml = 0; ml < 2; ++ml) {
        const int m = wave * 2 + ml;
#pragma unroll
        for (int nc = 0; nc < 4; ++nc) {
            const int col = m * 64 + nc * 16 + lrow;
            const float bv = bias[col];
            v8f d = acc[ml * 4 + nc];
#pragma unroll
            for (int q = 0; q < 8; ++q) {
                int b = q + lhal * 8;
                yt[b * OUT_F + col] = d[q] + bv;
            }
        }
    }
}

extern "C" void kernel_launch(void* const* d_in, const int* in_sizes, int n_in,
                              void* d_out, int out_size, void* d_ws, size_t ws_size,
                              hipStream_t stream) {
    (void)n_in; (void)out_size; (void)d_ws; (void)ws_size;
    const float* x    = (const float*)d_in[0];
    const float* c1   = (const float*)d_in[1];
    const float* c2   = (const float*)d_in[2];
    const float* bias = (const float*)d_in[3];
    float* out = (float*)d_out;

    const int rows = in_sizes[0] / IN_F;   // 16384
    const int grid = rows / TB;            // 1024 workgroups of 512 threads
    btt_fused_kernel<<<grid, THREADS, LDS_HALFS * sizeof(_Float16), stream>>>(
        x, c1, c2, bias, out);
}
